// Geoglobal_FPT_37855841747281
// MI455X (gfx1250) — matrix operand order repaired and checked by reading, not compiled, not run on hardware
//
#include <hip/hip_runtime.h>

// GeoGlobal attention, fused flash-style for MI455X (gfx1250, wave32).
// S = (2*QK^T - q2 - k2 + geo) * sqrt(D); softmax; O = W V.
// bf16 WMMA (v_wmma_f32_16x16x32_bf16) for both GEMMs, fp32 softmax/statistics.
// Round 3: __builtin_amdgcn_sched_barrier(0) fences keep all B-fragment DS
// loads issued BEFORE the WMMA chains (round-2 asm showed the scheduler sinking
// each load to its WMMA -> s_wait_dscnt 0x0 full drain per matrix op).

typedef __attribute__((ext_vector_type(16))) __bf16 v16bf;
typedef __attribute__((ext_vector_type(8)))  float  v8f;

namespace {
constexpr int BATCH = 16;
constexpr int SEQ   = 512;
constexpr int NHEAD = 8;
constexpr int HDIM  = 64;
constexpr int QT    = 64;   // query rows per workgroup (16 per wave, 4 waves)
constexpr int KTILE = 64;   // key tile
constexpr int LDST  = 68;   // padded LDS row stride in bf16 elems (even, avoids bank conflicts)
constexpr float LOG2E = 1.44269504088896340736f;
}

union BPack { unsigned u[8]; v16bf v; };

__device__ __forceinline__ __bf16 f2bf(float x) {
  // round-to-nearest-even f32 -> bf16, bit-level (no __bf16 arithmetic needed)
  unsigned u = __float_as_uint(x);
  u = (u + 0x7FFFu + ((u >> 16) & 1u)) >> 16;
  union { unsigned short s; __bf16 h; } c;
  c.s = (unsigned short)u;
  return c.h;
}

// A-matrix fragment (16-bit A 16x32 layout): lane = row (l16), lane-half selects
// K offset +0/+8; VGPR i<4 -> K = kl+2i, i>=4 -> K = kl+16+2(i-4). Pairs are
// contiguous in a row-major [row][k] LDS tile -> ds_load_b32/2addr_b64.
__device__ __forceinline__ v16bf load_a_frag(const __bf16* base, int row, int k0, int hh) {
  BPack p;
  const __bf16* rp = base + row * LDST;
  const int kl = k0 + (hh ? 8 : 0);
#pragma unroll
  for (int i = 0; i < 4; ++i)
    p.u[i] = *(const unsigned*)(rp + kl + 2 * i);
#pragma unroll
  for (int i = 0; i < 4; ++i)
    p.u[4 + i] = *(const unsigned*)(rp + kl + 16 + 2 * i);
  return p.v;
}

// B-matrix fragment (16-bit B 32x16 layout): lane = column N (l16), lane-half
// selects K offset +0/+8; VGPR j<4 holds K = kl+2j, j>=4 -> K = kl+16+2(j-4). Source tile
// must be [n][k] row-major so pairs are contiguous.
__device__ __forceinline__ v16bf load_b_frag(const __bf16* base, int n, int k0, int hh) {
  BPack p;
  const __bf16* np = base + n * LDST + k0 + (hh ? 8 : 0);
#pragma unroll
  for (int i = 0; i < 8; ++i)
    p.u[i] = *(const unsigned*)(np + 2 * i + (i >= 4 ? 8 : 0));
  return p.v;
}

// Reductions across the 16-lane group (C-layout row spans 16 lanes of wave32).
__device__ __forceinline__ float rmax16(float x) {
#pragma unroll
  for (int m = 1; m < 16; m <<= 1) x = fmaxf(x, __shfl_xor(x, m, 32));
  return x;
}
__device__ __forceinline__ float rsum16(float x) {
#pragma unroll
  for (int m = 1; m < 16; m <<= 1) x += __shfl_xor(x, m, 32);
  return x;
}

__global__ __launch_bounds__(128) void geo_attn_kernel(
    const float* __restrict__ q, const float* __restrict__ k,
    const float* __restrict__ v, const float* __restrict__ geo,
    float* __restrict__ out) {
  __shared__ __align__(16) __bf16 Qs[QT * LDST];      // [qrow][d]   bf16
  __shared__ __align__(16) __bf16 Ks[KTILE * LDST];   // [key][d]    bf16
  __shared__ __align__(16) __bf16 VTs[HDIM * LDST];   // [d][key]    bf16 (transposed)
  __shared__ __align__(16) __bf16 Ws[4 * 16 * LDST];  // per-wave P staging [16][64]
  __shared__ float q2s[QT];
  __shared__ float k2s[KTILE];
  __shared__ float psum[128];

  const int tid  = threadIdx.x;
  const int lane = tid & 31;
  const int wv   = tid >> 5;   // wave id 0..3
  const int hh   = lane >> 4;  // lane half (C-layout row selector)
  const int l16  = lane & 15;  // lane within half (C-layout column)

  const int qblk = blockIdx.x;
  const int hd   = blockIdx.y;
  const int bb   = blockIdx.z;

  const int qbase = qblk * QT;
  const size_t row_stride = (size_t)NHEAD * HDIM;  // 512 floats between rows (same head)
  const float* qbasep = q + (((size_t)bb * SEQ + qbase) * NHEAD + hd) * HDIM;
  const float* kbasep = k + ((size_t)bb * SEQ * NHEAD + hd) * HDIM;
  const float* vbasep = v + ((size_t)bb * SEQ * NHEAD + hd) * HDIM;

  // ---- Load Q tile (fp32 -> bf16 LDS) and fp32 row sums of squares ----
  {
    const int row = tid >> 1, seg = tid & 1;
    const float* sp = qbasep + (size_t)row * row_stride + seg * 32;
    __bf16* dp = Qs + row * LDST + seg * 32;
    float ss = 0.f;
#pragma unroll
    for (int i = 0; i < 32; ++i) { float x = sp[i]; ss += x * x; dp[i] = f2bf(x); }
    psum[tid] = ss;
  }
  __syncthreads();
  if (tid < QT) q2s[tid] = psum[2 * tid] + psum[2 * tid + 1];
  __syncthreads();

  // Persistent per-wave Q fragments (K=0..31, K=32..63) and q2 registers.
  const __bf16* qwave = Qs + wv * 16 * LDST;
  v16bf aq0 = load_a_frag(qwave, l16, 0, hh);
  v16bf aq1 = load_a_frag(qwave, l16, 32, hh);
  float q2r[8];
#pragma unroll
  for (int r = 0; r < 8; ++r) q2r[r] = q2s[wv * 16 + r + 8 * hh];

  float mrun[8], lrun[8];
#pragma unroll
  for (int r = 0; r < 8; ++r) { mrun[r] = -3.0e38f; lrun[r] = 0.f; }
  v8f o[4] = {};  // O accumulator: 4 column tiles x 8 f32 (C layout)

  __bf16* wsbase = Ws + wv * 16 * LDST;

  for (int kt = 0; kt < SEQ / KTILE; ++kt) {
    __syncthreads();  // previous iteration's Ks/VTs/k2s consumers done
    // ---- Load K tile (+sumsq partials) and V tile transposed ----
    {
      const int row = tid >> 1, seg = tid & 1;
      const float* sp = kbasep + (size_t)(kt * KTILE + row) * row_stride + seg * 32;
      __bf16* dp = Ks + row * LDST + seg * 32;
      float ss = 0.f;
#pragma unroll
      for (int i = 0; i < 32; ++i) { float x = sp[i]; ss += x * x; dp[i] = f2bf(x); }
      psum[tid] = ss;
      const float* vp = vbasep + (size_t)(kt * KTILE + row) * row_stride + seg * 32;
#pragma unroll
      for (int i = 0; i < 32; ++i) VTs[(seg * 32 + i) * LDST + row] = f2bf(vp[i]);
      if (kt + 1 < SEQ / KTILE) {  // global_prefetch_b8 of next tiles into cache
        __builtin_prefetch(sp + (size_t)KTILE * row_stride, 0, 3);
        __builtin_prefetch(vp + (size_t)KTILE * row_stride, 0, 3);
      }
    }
    __syncthreads();
    if (tid < KTILE) k2s[tid] = psum[2 * tid] + psum[2 * tid + 1];
    __syncthreads();

    // ---- S = Q K^T : issue ALL 16 DS loads, fence, then 8 WMMAs ----
    v16bf bk[8];
#pragma unroll
    for (int t = 0; t < 4; ++t) {
      bk[t]     = load_b_frag(Ks, t * 16 + l16, 0, hh);
      bk[4 + t] = load_b_frag(Ks, t * 16 + l16, 32, hh);
    }
    __builtin_amdgcn_sched_barrier(0);  // keep DS loads above the WMMA chain
    v8f acc[4] = {};
#pragma unroll
    for (int t = 0; t < 4; ++t)
      acc[t] = __builtin_amdgcn_wmma_f32_16x16x32_bf16(
          false, aq0, false, bk[t], (short)0, acc[t], false, false);
#pragma unroll
    for (int t = 0; t < 4; ++t)
      acc[t] = __builtin_amdgcn_wmma_f32_16x16x32_bf16(
          false, aq1, false, bk[4 + t], (short)0, acc[t], false, false);

    // ---- Preload V B-frags now: softmax VALU below hides their LDS latency --
    v16bf bv[8];
#pragma unroll
    for (int t = 0; t < 4; ++t) {
      bv[t]     = load_b_frag(VTs, t * 16 + l16, 0, hh);
      bv[4 + t] = load_b_frag(VTs, t * 16 + l16, 32, hh);
    }

    float k2c[4];
#pragma unroll
    for (int t = 0; t < 4; ++t) k2c[t] = k2s[t * 16 + l16];

    // ---- Online softmax (fp32), P -> bf16 LDS staging ----
#pragma unroll
    for (int r = 0; r < 8; ++r) {
      const int grow = qbase + wv * 16 + r + 8 * hh;
      const float* gp = geo + ((size_t)bb * SEQ + grow) * SEQ + kt * KTILE;
      float sv[4];
#pragma unroll
      for (int t = 0; t < 4; ++t) {
        float g = gp[t * 16 + l16];
        // total = (2*qk - q2 - k2 + geo) / D^-0.5 = (...) * 8
        sv[t] = (2.f * acc[t][r] - q2r[r] - k2c[t] + g) * 8.0f;
      }
      float mx = fmaxf(fmaxf(sv[0], sv[1]), fmaxf(sv[2], sv[3]));
      mx = rmax16(mx);
      const float mnew  = fmaxf(mrun[r], mx);
      const float alpha = exp2f((mrun[r] - mnew) * LOG2E);
      mrun[r] = mnew;
      float ps = 0.f;
#pragma unroll
      for (int t = 0; t < 4; ++t) {
        float p = exp2f((sv[t] - mnew) * LOG2E);
        ps += p;
        wsbase[(r + 8 * hh) * LDST + t * 16 + l16] = f2bf(p);
      }
      ps = rsum16(ps);
      lrun[r] = lrun[r] * alpha + ps;
#pragma unroll
      for (int t = 0; t < 4; ++t) o[t][r] *= alpha;
    }

    // ---- O += P V : wave-private W (LDS in-order per wave, no barrier) ----
    v16bf aw0 = load_a_frag(wsbase, l16, 0, hh);
    v16bf aw1 = load_a_frag(wsbase, l16, 32, hh);
    __builtin_amdgcn_sched_barrier(0);  // aw/bv loads above, WMMA block below
#pragma unroll
    for (int t = 0; t < 4; ++t)
      o[t] = __builtin_amdgcn_wmma_f32_16x16x32_bf16(
          false, aw0, false, bv[t], (short)0, o[t], false, false);
#pragma unroll
    for (int t = 0; t < 4; ++t)
      o[t] = __builtin_amdgcn_wmma_f32_16x16x32_bf16(
          false, aw1, false, bv[4 + t], (short)0, o[t], false, false);
  }

  // ---- Epilogue: O /= l, scatter back to [B,N,H,D] ----
#pragma unroll
  for (int r = 0; r < 8; ++r) {
    const float invl = 1.0f / lrun[r];
    const int nrow = qbase + wv * 16 + r + 8 * hh;
    float* op = out + (((size_t)bb * SEQ + nrow) * NHEAD + hd) * HDIM;
#pragma unroll
    for (int t = 0; t < 4; ++t) op[t * 16 + l16] = o[t][r] * invl;
  }
}

extern "C" void kernel_launch(void* const* d_in, const int* in_sizes, int n_in,
                              void* d_out, int out_size, void* d_ws, size_t ws_size,
                              hipStream_t stream) {
  (void)in_sizes; (void)n_in; (void)d_ws; (void)ws_size; (void)out_size;
  const float* q   = (const float*)d_in[0];
  const float* k   = (const float*)d_in[1];
  const float* v   = (const float*)d_in[2];
  const float* geo = (const float*)d_in[3];
  float* out = (float*)d_out;

  dim3 grid(SEQ / QT, NHEAD, BATCH);  // (8, 8, 16) = 1024 workgroups
  dim3 block(128);                    // 4 waves of 32
  geo_attn_kernel<<<grid, block, 0, stream>>>(q, k, v, geo, out);
}
